// ImprovedSAE_46059229282444
// MI455X (gfx1250) — matrix operand order repaired and checked by reading, not compiled, not run on hardware
//
#include <hip/hip_runtime.h>

// ---------------------------------------------------------------------------
// Top-k SAE forward for MI455X (gfx1250, wave32, WMMA).
//   h     = relu(x @ W_enc^T + b_enc)        [32768 x 2048]
//   h     = topk-mask(h, k=64) per row
//   recon = h @ W_dec^T                      [32768 x 768]
// GEMMs on V_WMMA_F32_16X16X32_BF16 (bf16 in, f32 accumulate).
// All operands are L2-resident (192 MB), fragments load straight from global.
//
// Two data paths chosen by ws_size in kernel_launch:
//   FULL  (ws >= ~191 MB): x and masked-h pre-converted to bf16 in ws; the
//          GEMM inner loops are pure global_load_b128 + v_wmma.
//   SMALL (ws >=  6.3 MB): weights-only in ws; A operand converted f32->bf16
//          on the fly with packed v_cvt where available.
// ---------------------------------------------------------------------------

typedef __bf16 bf16;
typedef __attribute__((ext_vector_type(16))) __bf16 v16bf;
typedef __attribute__((ext_vector_type(8)))  __bf16 v8bf;
typedef __attribute__((ext_vector_type(4)))  __bf16 v4bf;
typedef __attribute__((ext_vector_type(2)))  __bf16 v2bf;
typedef __attribute__((ext_vector_type(8)))  float  v8f;
typedef __attribute__((ext_vector_type(4)))  float  v4f;

#define NROWS 32768
#define DIN   768
#define DHID  2048

// --------------------------- f32 -> bf16 helpers ---------------------------
// scalar RNE fallback (3 VALU/elem)
__device__ inline bf16 f2bf(float f) {
  unsigned u = __float_as_uint(f);
  unsigned r = u + 0x7FFFu + ((u >> 16) & 1u);
  unsigned short s = (unsigned short)(r >> 16);
  return __builtin_bit_cast(bf16, s);
}

// packed pair conversion: 1 VALU op for 2 elements when the HW op exists
__device__ inline v2bf f2bf2(float f0, float f1) {
#if __has_builtin(__builtin_amdgcn_cvt_pk_bf16_f32)
  auto t = __builtin_amdgcn_cvt_pk_bf16_f32(f0, f1);
  return __builtin_bit_cast(v2bf, t);
#else
  v2bf r;
  r[0] = f2bf(f0);
  r[1] = f2bf(f1);
  return r;
#endif
}

__device__ inline v8bf cvt8(v4f a, v4f b) {
  v8bf o;
  v2bf t;
  t = f2bf2(a[0], a[1]); o[0] = t[0]; o[1] = t[1];
  t = f2bf2(a[2], a[3]); o[2] = t[0]; o[3] = t[1];
  t = f2bf2(b[0], b[1]); o[4] = t[0]; o[5] = t[1];
  t = f2bf2(b[2], b[3]); o[6] = t[0]; o[7] = t[1];
  return o;
}

// ---------------------------------------------------------------------------
// Fragment loaders for V_WMMA_F32_16X16X32_BF16 (CDNA5 ISA §7.12.2).
//
// A (16x32 MxK) per-lane v16bf:
//   lanes 0-15 : M=lane,    elems 0-7 = K 0..7,  elems 8-15 = K 16..23
//   lanes 16-31: M=lane-16, elems 0-7 = K 8..15, elems 8-15 = K 24..31
// => two contiguous 16B bf16 chunks per lane from row-major [M x K].
__device__ inline v16bf load_a_frag_bf16(const bf16* __restrict__ base, int ld,
                                         int m0, int k0, int lane) {
  int half = lane >> 4;
  int row  = m0 + (lane & 15);
  const bf16* p = base + (size_t)row * ld + k0 + half * 8;
  v8bf lo = *(const v8bf*)(p);       // K = k0 + half*8 .. +7
  v8bf hi = *(const v8bf*)(p + 16);  // K = k0+16 + half*8 .. +7
  return __builtin_shufflevector(lo, hi, 0, 1, 2, 3, 4, 5, 6, 7,
                                 8, 9, 10, 11, 12, 13, 14, 15);
}

// same fragment, but from an f32 source with on-the-fly packed conversion
__device__ inline v16bf load_a_frag_f32(const float* __restrict__ base, int ld,
                                        int m0, int k0, int lane) {
  int half = lane >> 4;
  int row  = m0 + (lane & 15);
  const float* p0 = base + (size_t)row * ld + k0 + half * 8;
  v4f a0 = *(const v4f*)(p0);
  v4f a1 = *(const v4f*)(p0 + 4);
  v4f b0 = *(const v4f*)(p0 + 16);
  v4f b1 = *(const v4f*)(p0 + 20);
  v8bf lo = cvt8(a0, a1);
  v8bf hi = cvt8(b0, b1);
  return __builtin_shufflevector(lo, hi, 0, 1, 2, 3, 4, 5, 6, 7,
                                 8, 9, 10, 11, 12, 13, 14, 15);
}

// B (32x16 KxN) per-lane v16bf, same K order as A: lanes 0-15: col=lane,
// elems 0-7 = K 0..7, 8-15 = K 16..23; lanes 16-31: K 8..15, 24..31.
// Source: row-major [Ncol x K] (= B^T), i.e. W_enc / W_dec as stored.
__device__ inline v16bf load_b_frag_bf16(const bf16* __restrict__ base, int ld,
                                         int n0, int k0, int lane) {
  int khalf = lane >> 4;
  int col   = n0 + (lane & 15);
  const bf16* p = base + (size_t)col * ld + k0 + khalf * 8;
  v8bf lo = *(const v8bf*)(p);
  v8bf hi = *(const v8bf*)(p + 16);
  return __builtin_shufflevector(lo, hi, 0, 1, 2, 3, 4, 5, 6, 7,
                                 8, 9, 10, 11, 12, 13, 14, 15);
}

// ---------------------------------------------------------------------------
// GEMM: C[M x Ncol] = op(A[M x K] @ Bt[Ncol x K]^T), Bt bf16, C f32.
// 128x128 workgroup tile, 256 threads = 8 waves in a 2(M) x 4(N) grid;
// each wave owns 4x2 16x16 tiles (A frags reused 2x, B frags 4x).
// ---------------------------------------------------------------------------
template <bool BIAS_RELU, bool A_IS_BF16>
__global__ __launch_bounds__(256) void sae_gemm_kernel(
    const void* __restrict__ Av,     // [M x K] row-major, f32 or bf16
    const bf16* __restrict__ Bt,     // [Ncol x K] bf16 row-major
    const float* __restrict__ bias,  // [Ncol] (encoder only)
    float* __restrict__ C,           // [M x Ncol] f32 row-major
    int K, int Ncol) {
  const int wave = threadIdx.x >> 5;
  const int lane = threadIdx.x & 31;
  const int wm = wave >> 2;                    // 0..1
  const int wn = wave & 3;                     // 0..3
  const int m0 = blockIdx.y * 128 + wm * 64;   // 4 M-tiles
  const int n0 = blockIdx.x * 128 + wn * 32;   // 2 N-tiles

  v8f acc[4][2];
#pragma unroll
  for (int i = 0; i < 4; ++i)
#pragma unroll
    for (int j = 0; j < 2; ++j)
      acc[i][j] = (v8f){0.f, 0.f, 0.f, 0.f, 0.f, 0.f, 0.f, 0.f};

  for (int k0 = 0; k0 < K; k0 += 32) {
    v16bf a[4], b[2];
#pragma unroll
    for (int i = 0; i < 4; ++i) {
      if constexpr (A_IS_BF16)
        a[i] = load_a_frag_bf16((const bf16*)Av, K, m0 + i * 16, k0, lane);
      else
        a[i] = load_a_frag_f32((const float*)Av, K, m0 + i * 16, k0, lane);
    }
#pragma unroll
    for (int j = 0; j < 2; ++j)
      b[j] = load_b_frag_bf16(Bt, K, n0 + j * 16, k0, lane);
#pragma unroll
    for (int i = 0; i < 4; ++i)
#pragma unroll
      for (int j = 0; j < 2; ++j)
        acc[i][j] = __builtin_amdgcn_wmma_f32_16x16x32_bf16(
            /*neg_a=*/false, a[i], /*neg_b=*/false, b[j],
            /*c_mod=*/(short)0, acc[i][j], /*reuse_a=*/false, /*reuse_b=*/false);
  }

  // C/D layout: VGPR r -> row = r + 8*(lane>>4), col = lane&15 (per tile)
  const int colLane = lane & 15;
  const int rowHalf = (lane >> 4) * 8;
#pragma unroll
  for (int j = 0; j < 2; ++j) {
    const int col = n0 + j * 16 + colLane;
    float bv = 0.f;
    if constexpr (BIAS_RELU) bv = bias[col];
#pragma unroll
    for (int i = 0; i < 4; ++i) {
#pragma unroll
      for (int r = 0; r < 8; ++r) {
        const int row = m0 + i * 16 + rowHalf + r;
        float v = acc[i][j][r];
        if constexpr (BIAS_RELU) {
          v += bv;
          v = v > 0.f ? v : 0.f;
        }
        C[(size_t)row * Ncol + col] = v;
      }
    }
  }
}

// ---------------------------------------------------------------------------
// Weight conversion: f32 -> bf16 for W_enc [2048x768] and W_dec [768x2048].
// ---------------------------------------------------------------------------
__global__ __launch_bounds__(256) void convert_weights_kernel(
    const float* __restrict__ Wenc, const float* __restrict__ Wdec,
    bf16* __restrict__ enc_out, bf16* __restrict__ dec_out) {
  const int PER = DHID * DIN / 8;  // 196608 groups of 8 per matrix
  int idx = blockIdx.x * 256 + threadIdx.x;
  const float* src;
  bf16* dst;
  if (idx < PER) {
    src = Wenc; dst = enc_out;
  } else {
    src = Wdec; dst = dec_out; idx -= PER;
  }
  const float* p = src + (size_t)idx * 8;
  v4f a = *(const v4f*)p;
  v4f b = *(const v4f*)(p + 4);
  *(v8bf*)(dst + (size_t)idx * 8) = cvt8(a, b);
}

// x conversion: f32 -> bf16, [32768 x 768]
__global__ __launch_bounds__(256) void convert_x_kernel(
    const float* __restrict__ x, bf16* __restrict__ xb) {
  size_t idx = (size_t)blockIdx.x * 256 + threadIdx.x;  // 8 floats each
  const float* p = x + idx * 8;
  v4f a = *(const v4f*)p;
  v4f b = *(const v4f*)(p + 4);
  *(v8bf*)(xb + idx * 8) = cvt8(a, b);
}

// ---------------------------------------------------------------------------
// Per-row exact top-k mask via 4-pass radix select over float bit patterns
// (ReLU output >= 0 -> unsigned bit pattern is order-preserving).
// One 256-thread block per row, 8 values/thread, 256-bin LDS histogram.
// Optionally emits a bf16 copy of the masked row for the decoder (FULL path).
// ---------------------------------------------------------------------------
__global__ __launch_bounds__(256) void topk_mask_kernel(
    float* __restrict__ h, const int* __restrict__ kptr,
    bf16* __restrict__ hbf /* nullptr in SMALL path */) {
  __shared__ unsigned bins[256];
  __shared__ unsigned sh_prefix;
  __shared__ int sh_rem;
  __shared__ int sh_take;
  float* row = h + (size_t)blockIdx.x * DHID;
  const int tid = threadIdx.x;

  float v[8];
  unsigned b[8];
  v4f lo = *(const v4f*)(row + tid * 8);
  v4f hi = *(const v4f*)(row + tid * 8 + 4);
#pragma unroll
  for (int i = 0; i < 4; ++i) { v[i] = lo[i]; v[i + 4] = hi[i]; }
#pragma unroll
  for (int i = 0; i < 8; ++i) b[i] = __float_as_uint(v[i]);

  if (tid == 0) { sh_prefix = 0u; sh_rem = *kptr; sh_take = 0; }

  for (int pass = 0; pass < 4; ++pass) {
    const int shift = 24 - pass * 8;
    bins[tid] = 0u;
    __syncthreads();
    const unsigned known  = pass ? (0xFFFFFFFFu << (shift + 8)) : 0u;
    const unsigned prefix = sh_prefix;
#pragma unroll
    for (int i = 0; i < 8; ++i)
      if ((b[i] & known) == prefix) atomicAdd(&bins[(b[i] >> shift) & 255u], 1u);
    __syncthreads();
    if (tid == 0) {
      int rem = sh_rem, cum = 0, d = 0;
      for (int q = 255; q >= 0; --q) {
        int c = (int)bins[q];
        if (cum + c >= rem) { d = q; sh_rem = rem - cum; break; }
        cum += c;
      }
      sh_prefix = prefix | ((unsigned)d << shift);
    }
    __syncthreads();
  }

  const unsigned T = sh_prefix;  // bit pattern of k-th largest value
  const int need_eq = sh_rem;    // how many ==T elements still to keep
  float o[8];
#pragma unroll
  for (int i = 0; i < 8; ++i) {
    float val = 0.f;
    if (b[i] > T) {
      val = v[i];
    } else if (b[i] == T) {
      int p = atomicAdd(&sh_take, 1);
      if (p < need_eq) val = v[i];
    }
    o[i] = val;
  }
  v4f olo, ohi;
#pragma unroll
  for (int i = 0; i < 4; ++i) { olo[i] = o[i]; ohi[i] = o[i + 4]; }
  *(v4f*)(row + tid * 8) = olo;
  *(v4f*)(row + tid * 8 + 4) = ohi;

  if (hbf) {  // bf16 masked copy for the decoder A operand (FULL path)
    *(v8bf*)(hbf + (size_t)blockIdx.x * DHID + tid * 8) = cvt8(olo, ohi);
  }
}

// ---------------------------------------------------------------------------
// kernel_launch
// Inputs: x [N*DIN] f32, W_enc [DHID*DIN] f32, b_enc [DHID] f32,
//         W_dec [DIN*DHID] f32, k int (1 elem)
// Output: recon [N*DIN] f32 then h [N*DHID] f32, concatenated.
// ---------------------------------------------------------------------------
extern "C" void kernel_launch(void* const* d_in, const int* in_sizes, int n_in,
                              void* d_out, int out_size, void* d_ws, size_t ws_size,
                              hipStream_t stream) {
  const float* x    = (const float*)d_in[0];
  const float* Wenc = (const float*)d_in[1];
  const float* benc = (const float*)d_in[2];
  const float* Wdec = (const float*)d_in[3];
  const int*   kptr = (const int*)d_in[4];

  float* recon = (float*)d_out;                        // [N x DIN]
  float* h     = (float*)d_out + (size_t)NROWS * DIN;  // [N x DHID]

  // workspace layout (byte offsets; all sizes 256B-multiples)
  const size_t sz_wenc = (size_t)DHID * DIN * sizeof(bf16);
  const size_t sz_wdec = (size_t)DIN * DHID * sizeof(bf16);
  const size_t sz_xbf  = (size_t)NROWS * DIN * sizeof(bf16);
  const size_t sz_hbf  = (size_t)NROWS * DHID * sizeof(bf16);
  char* ws = (char*)d_ws;
  bf16* wenc_bf = (bf16*)(ws);
  bf16* wdec_bf = (bf16*)(ws + sz_wenc);
  bf16* x_bf    = (bf16*)(ws + sz_wenc + sz_wdec);
  bf16* h_bf    = (bf16*)(ws + sz_wenc + sz_wdec + sz_xbf);
  const bool full = ws_size >= sz_wenc + sz_wdec + sz_xbf + sz_hbf;

  // 1) weights f32 -> bf16
  convert_weights_kernel<<<dim3(2 * (DHID * DIN / 8) / 256), dim3(256), 0, stream>>>(
      Wenc, Wdec, wenc_bf, wdec_bf);

  if (full) {
    // 1b) x f32 -> bf16 once (encoder re-reads x 16x; convert once, not 16x)
    convert_x_kernel<<<dim3((NROWS * DIN / 8) / 256), dim3(256), 0, stream>>>(x, x_bf);

    // 2) encoder: pure bf16-load + wmma inner loop
    sae_gemm_kernel<true, true><<<dim3(DHID / 128, NROWS / 128), dim3(256), 0, stream>>>(
        x_bf, wenc_bf, benc, h, DIN, DHID);

    // 3) top-k mask, also emitting bf16 masked h
    topk_mask_kernel<<<dim3(NROWS), dim3(256), 0, stream>>>(h, kptr, h_bf);

    // 4) decoder: pure bf16-load + wmma inner loop
    sae_gemm_kernel<false, true><<<dim3(DIN / 128, NROWS / 128), dim3(256), 0, stream>>>(
        h_bf, wdec_bf, nullptr, recon, DHID, DIN);
  } else {
    // fallback: convert A on the fly (packed v_cvt), ws holds weights only
    sae_gemm_kernel<true, false><<<dim3(DHID / 128, NROWS / 128), dim3(256), 0, stream>>>(
        x, wenc_bf, benc, h, DIN, DHID);
    topk_mask_kernel<<<dim3(NROWS), dim3(256), 0, stream>>>(h, kptr, nullptr);
    sae_gemm_kernel<false, false><<<dim3(DIN / 128, NROWS / 128), dim3(256), 0, stream>>>(
        h, wdec_bf, nullptr, recon, DHID, DIN);
  }
}
